// SingleHeadAttention_13726715478116
// MI455X (gfx1250) — matrix operand order repaired and checked by reading, not compiled, not run on hardware
//
#include <hip/hip_runtime.h>
#include <math.h>

typedef __bf16 bf16;
typedef __attribute__((ext_vector_type(16))) __bf16 v16bf;
typedef __attribute__((ext_vector_type(8)))  float  v8f;

#define DEVFN static __device__ __forceinline__

// ---------------- problem sizes (hardcoded from reference) ----------------
#define BB 8
#define SS 2048
#define TT 2048
#define DD 1024

// ---------------- workspace layout (bytes) ----------------
static const size_t OFF_GATES = 0;                         // gq[1024], gk[1024], gv[1024] f32
static const size_t OFF_WQB   = 16ull << 10;               // D*D bf16          (2 MB)
static const size_t OFF_XB    = OFF_WQB + (2ull  << 20);   // B*S*D bf16        (32 MB)
static const size_t OFF_QB    = OFF_XB  + (32ull << 20);   // B*S*D bf16        (32 MB)
static const size_t OFF_KB    = OFF_QB  + (32ull << 20);   // B*T*D bf16        (32 MB)
static const size_t OFF_VB    = OFF_KB  + (32ull << 20);   // B*T*D bf16        (32 MB)
static const size_t OFF_QLIN  = OFF_VB  + (32ull << 20);   // B*S*D f32 (reused as attn-out) (64 MB)
static const size_t OFF_P     = OFF_QLIN+ (64ull << 20);   // B*S*T f32 scores / bf16 p      (128 MB)

DEVFN float sigmoidf_(float x) { return 1.0f / (1.0f + __expf(-x)); }

// ---------------- CDNA5 async copy: global -> LDS (ASYNCcnt-tracked) ----------------
// LDS byte offset = low 32 bits of the flat shared pointer (AS3->flat cast puts the
// LDS offset in addr[31:0]; hardware maps flat-LDS addresses by truncation, ISA 10.2).
DEVFN unsigned lds_off(const void* p) { return (unsigned)(unsigned long long)p; }

DEVFN void async_b128(unsigned lds, const void* g) {
  asm volatile("global_load_async_to_lds_b128 %0, %1, off" :: "v"(lds), "v"(g) : "memory");
}

DEVFN void wait_async_le(int n) {
#if __has_builtin(__builtin_amdgcn_s_wait_asynccnt)
  if (n == 0) __builtin_amdgcn_s_wait_asynccnt(0);
  else        __builtin_amdgcn_s_wait_asynccnt(4);
#else
  if (n == 0) asm volatile("s_wait_asynccnt 0" ::: "memory");
  else        asm volatile("s_wait_asynccnt 4" ::: "memory");
#endif
}

// ---------------- WMMA fragment loaders (16x16x32 bf16) ----------------
// LDS pitches chosen so every fragment read / async b128 LDS write is 16B aligned.
#define LDKA 40   // [row][k] tiles: 128 x 32 halves, pitch 40 (80B = 16B*5, bank spread)
#define LDNK 136  // [k][n] tile: 32 x 128 halves, pitch 136 (272B = 16B*17)

// A 16x32 (MxK), LDS row-major [m][k]: lane m=lane&15, g=lane>>4;
//   e -> K=16g+e
DEVFN v16bf loadA_frag(const bf16* t, int ld, int lane) {
  int m = lane & 15, g = lane >> 4;
  const bf16* p = t + m * ld + 16 * g;
  v16bf r;
#pragma unroll
  for (int e = 0; e < 8; ++e) r[e] = p[e];
#pragma unroll
  for (int e = 0; e < 8; ++e) r[8 + e] = p[8 + e];
  return r;
}
// B 32x16 (KxN) from LDS stored [n][k]: lane n=lane&15, g=lane>>4; e -> K=16g+e
DEVFN v16bf loadB_frag(const bf16* t, int ld, int lane) {
  int n = lane & 15, g = lane >> 4;
  const bf16* p = t + n * ld + 16 * g;
  v16bf r;
#pragma unroll
  for (int e = 0; e < 16; ++e) r[e] = p[e];
  return r;
}
// B 32x16 (KxN) from LDS stored [k][n]: lane n=lane&15, g=lane>>4; e -> K=16g+e
DEVFN v16bf loadBkn_frag(const bf16* t, int ld, int lane) {
  int n = lane & 15, g = lane >> 4;
  v16bf r;
#pragma unroll
  for (int e = 0; e < 16; ++e) r[e] = t[(16 * g + e) * ld + n];
  return r;
}

// Issue one K-step tile (A: 128x32, B: 128x32 or 32x128) as async b128 copies.
// 512 x 16B ops per array over 256 threads -> 2 instructions/thread/array
// -> per wave per tile: 2 (A) + 2 (B) = 4 ASYNCcnt increments.
template <bool B_IS_KN>
DEVFN void issue_tile(const bf16* Ag, const bf16* Bg, int lda, int ldb,
                      int m0, int n0, int k0, bf16* Al, bf16* Bl, int tid) {
#pragma unroll
  for (int j = 0; j < 2; ++j) {
    int o = tid + 256 * j;
    int row = o >> 2, seg = (o & 3) * 8;
    async_b128(lds_off(Al + row * LDKA + seg),
               Ag + (long long)(m0 + row) * lda + k0 + seg);
  }
  if (B_IS_KN) {
#pragma unroll
    for (int j = 0; j < 2; ++j) {
      int o = tid + 256 * j;
      int row = o >> 4, seg = (o & 15) * 8;   // row = k (0..31), seg along n
      async_b128(lds_off(Bl + row * LDNK + seg),
                 Bg + (long long)(k0 + row) * ldb + n0 + seg);
    }
  } else {
#pragma unroll
    for (int j = 0; j < 2; ++j) {
      int o = tid + 256 * j;
      int row = o >> 2, seg = (o & 3) * 8;    // row = n (0..127), seg along k
      async_b128(lds_off(Bl + row * LDKA + seg),
                 Bg + (long long)(n0 + row) * ldb + k0 + seg);
    }
  }
}

// ---------------- tiled WMMA GEMM: C = scale*(A @ op(B)) + bias ----------------
// A: [M][K] bf16 row-major. B_IS_KN ? B:[K][N] : B:[N][K]. C: f32 [M][ldc].
// Block: 256 thr = 8 waves (4 along M x 2 along N), tile 128x128, K-step 32.
// Double-buffered LDS, async global->LDS copies overlapped with WMMA compute.
template <bool B_IS_KN>
__global__ void __launch_bounds__(256)
wmma_gemm(const bf16* __restrict__ Ag, const bf16* __restrict__ Bg,
          float* __restrict__ Cg, int K, int lda, int ldb, int ldc,
          long long sa, long long sb, long long sc,
          const float* __restrict__ bias, float scale) {
  __shared__ alignas(16) bf16 Al[2][128 * LDKA];
  __shared__ alignas(16) bf16 Bl[2][128 * LDKA];   // kn layout uses 32*LDNK = 4352 <= 5120

  int bz = blockIdx.z;
  Ag += (long long)bz * sa;
  Bg += (long long)bz * sb;
  Cg += (long long)bz * sc;

  int tid = threadIdx.x;
  int lane = tid & 31, wid = tid >> 5;
  int wm = wid & 3, wn = wid >> 2;  // 4 x 2 wave grid
  int m0 = blockIdx.y * 128, n0 = blockIdx.x * 128;

  v8f acc[2][4];
#pragma unroll
  for (int i = 0; i < 2; i++)
#pragma unroll
    for (int j = 0; j < 4; j++)
#pragma unroll
      for (int v = 0; v < 8; v++) acc[i][j][v] = 0.0f;

  int nk = K >> 5;
  issue_tile<B_IS_KN>(Ag, Bg, lda, ldb, m0, n0, 0, Al[0], Bl[0], tid);

  for (int i = 0; i < nk; ++i) {
    int cur = i & 1;
    if (i + 1 < nk) {
      // prefetch next tile into the other buffer while this one computes
      issue_tile<B_IS_KN>(Ag, Bg, lda, ldb, m0, n0, (i + 1) << 5,
                          Al[cur ^ 1], Bl[cur ^ 1], tid);
      wait_async_le(4);   // tile i landed (only tile i+1's 4 ops outstanding)
    } else {
      wait_async_le(0);
    }
    __syncthreads();      // all waves' tile-i data visible in LDS

    v16bf af[2], bfr[4];
#pragma unroll
    for (int j = 0; j < 2; j++)
      af[j] = loadA_frag(Al[cur] + (wm * 32 + j * 16) * LDKA, LDKA, lane);
    if (B_IS_KN) {
#pragma unroll
      for (int j = 0; j < 4; j++)
        bfr[j] = loadBkn_frag(Bl[cur] + (wn * 64 + j * 16), LDNK, lane);
    } else {
#pragma unroll
      for (int j = 0; j < 4; j++)
        bfr[j] = loadB_frag(Bl[cur] + (wn * 64 + j * 16) * LDKA, LDKA, lane);
    }

#pragma unroll
    for (int mi = 0; mi < 2; mi++)
#pragma unroll
      for (int nj = 0; nj < 4; nj++)
        acc[mi][nj] = __builtin_amdgcn_wmma_f32_16x16x32_bf16(
            false, af[mi], false, bfr[nj], (short)0, acc[mi][nj], false, false);

    __syncthreads();      // compute on buffer cur done before it is overwritten
  }

  // --- epilogue: C frag element (v,lane): m = v + 8*(lane>>4), n = lane&15 ---
  int g = lane >> 4, nn = lane & 15;
#pragma unroll
  for (int i = 0; i < 2; i++)
#pragma unroll
    for (int j = 0; j < 4; j++) {
      int col = n0 + wn * 64 + j * 16 + nn;
      float bb = bias ? bias[col] : 0.0f;
#pragma unroll
      for (int v = 0; v < 8; v++) {
        int row = m0 + wm * 32 + i * 16 + v + 8 * g;
        Cg[(long long)row * ldc + col] = acc[i][j][v] * scale + bb;
      }
    }
}

// ---------------- gates: gq=sig(qs), gk=sig(ks), gv=sig(a)*tanh(b), ab=Wov@sig(vs)+bov ----
__global__ void __launch_bounds__(256)
gates_kernel(const float* __restrict__ qs, const float* __restrict__ ks_,
             const float* __restrict__ vs, const float* __restrict__ Wov,
             const float* __restrict__ bov, float* __restrict__ gq,
             float* __restrict__ gk, float* __restrict__ gv) {
  __shared__ float sv[DD];
  int tid = threadIdx.x;
  int d = blockIdx.x * 256 + tid;
  for (int i = tid; i < DD; i += 256) sv[i] = sigmoidf_(vs[i]);
  __syncthreads();
  float a = bov[d], bg = bov[DD + d];
  const float* wa = Wov + (long long)d * DD;
  const float* wb = Wov + (long long)(DD + d) * DD;
  for (int j = 0; j < DD; ++j) {
    a += wa[j] * sv[j];
    bg += wb[j] * sv[j];
  }
  gv[d] = sigmoidf_(a) * tanhf(bg);
  gq[d] = sigmoidf_(qs[d]);
  gk[d] = sigmoidf_(ks_[d]);
}

// ---------------- elementwise conversions ----------------
__global__ void __launch_bounds__(256)
f32_to_bf16_kernel(const float* __restrict__ in, bf16* __restrict__ out, long long n) {
  long long i = (long long)blockIdx.x * 256 + threadIdx.x;
  if (i < n) out[i] = (bf16)in[i];
}

__global__ void __launch_bounds__(256)
gate_convert_y(const float* __restrict__ Y, const float* __restrict__ gk,
               const float* __restrict__ gv, bf16* __restrict__ kb,
               bf16* __restrict__ vb) {
  long long i = (long long)blockIdx.x * 256 + threadIdx.x;
  int d = (int)(i & (DD - 1));
  float y = Y[i];
  kb[i] = (bf16)(y * gk[d]);
  vb[i] = (bf16)(y * gv[d]);
}

// ---------------- LayerNorm(qlin)*gq -> bf16 q (one block per row of 1024) ----
__global__ void __launch_bounds__(256)
ln_gate_to_bf16(const float* __restrict__ in, bf16* __restrict__ out,
                const float* __restrict__ gam, const float* __restrict__ bet,
                const float* __restrict__ gate) {
  __shared__ float red[256];
  long long r = blockIdx.x;
  int tid = threadIdx.x;
  const float* row = in + r * DD;
  float x[4];
  float s = 0.f;
#pragma unroll
  for (int i = 0; i < 4; i++) { x[i] = row[tid + i * 256]; s += x[i]; }
  red[tid] = s; __syncthreads();
  for (int off = 128; off; off >>= 1) { if (tid < off) red[tid] += red[tid + off]; __syncthreads(); }
  float mean = red[0] * (1.0f / DD); __syncthreads();
  float v = 0.f;
#pragma unroll
  for (int i = 0; i < 4; i++) { float dlt = x[i] - mean; v += dlt * dlt; }
  red[tid] = v; __syncthreads();
  for (int off = 128; off; off >>= 1) { if (tid < off) red[tid] += red[tid + off]; __syncthreads(); }
  float rstd = rsqrtf(red[0] * (1.0f / DD) + 1e-5f);
#pragma unroll
  for (int i = 0; i < 4; i++) {
    int c = tid + i * 256;
    out[r * DD + c] = (bf16)(((x[i] - mean) * rstd * gam[c] + bet[c]) * gate[c]);
  }
}

// ---------------- softmax row (T=2048), mask applied, bf16 p written in place ----
__global__ void __launch_bounds__(256)
softmax_rows(float* __restrict__ scores, const int* __restrict__ mask) {
  __shared__ float red[256];
  long long r = blockIdx.x;          // b*S + s
  int s = (int)(r & (SS - 1));
  float* row = scores + r * TT;
  const int* mrow = mask + (long long)s * TT;
  int tid = threadIdx.x;
  float x[8];
  float mx = -__builtin_inff();
#pragma unroll
  for (int i = 0; i < 8; i++) {
    int t = tid + i * 256;
    float v = row[t];
    if (mrow[t] == 0) v = -__builtin_inff();
    x[i] = v;
    mx = fmaxf(mx, v);
  }
  red[tid] = mx; __syncthreads();
  for (int off = 128; off; off >>= 1) { if (tid < off) red[tid] = fmaxf(red[tid], red[tid + off]); __syncthreads(); }
  mx = red[0]; __syncthreads();
  float sum = 0.f;
#pragma unroll
  for (int i = 0; i < 8; i++) { x[i] = __expf(x[i] - mx); sum += x[i]; }
  red[tid] = sum; __syncthreads();
  for (int off = 128; off; off >>= 1) { if (tid < off) red[tid] += red[tid + off]; __syncthreads(); }
  float inv = 1.0f / red[0];
  __syncthreads();                   // all row reads complete before in-place bf16 writes
  bf16* prow = (bf16*)scores + r * (2 * TT);   // first half of the f32 row span holds bf16 p
#pragma unroll
  for (int i = 0; i < 8; i++) {
    int t = tid + i * 256;
    prow[t] = (bf16)(x[i] * inv);
  }
}

// ---------------- final LayerNorm(X + attn) -> f32 out ----------------
__global__ void __launch_bounds__(256)
add_ln_kernel(const float* __restrict__ X, const float* __restrict__ attn,
              const float* __restrict__ gam, const float* __restrict__ bet,
              float* __restrict__ out) {
  __shared__ float red[256];
  long long r = blockIdx.x;
  int tid = threadIdx.x;
  const float* xr = X + r * DD;
  const float* ar = attn + r * DD;
  float x[4];
  float s = 0.f;
#pragma unroll
  for (int i = 0; i < 4; i++) { int c = tid + i * 256; x[i] = xr[c] + ar[c]; s += x[i]; }
  red[tid] = s; __syncthreads();
  for (int off = 128; off; off >>= 1) { if (tid < off) red[tid] += red[tid + off]; __syncthreads(); }
  float mean = red[0] * (1.0f / DD); __syncthreads();
  float v = 0.f;
#pragma unroll
  for (int i = 0; i < 4; i++) { float dlt = x[i] - mean; v += dlt * dlt; }
  red[tid] = v; __syncthreads();
  for (int off = 128; off; off >>= 1) { if (tid < off) red[tid] += red[tid + off]; __syncthreads(); }
  float rstd = rsqrtf(red[0] * (1.0f / DD) + 1e-5f);
#pragma unroll
  for (int i = 0; i < 4; i++) {
    int c = tid + i * 256;
    out[r * DD + c] = (x[i] - mean) * rstd * gam[c] + bet[c];
  }
}

// ---------------- launcher ----------------
extern "C" void kernel_launch(void* const* d_in, const int* in_sizes, int n_in,
                              void* d_out, int out_size, void* d_ws, size_t ws_size,
                              hipStream_t stream) {
  const float* X   = (const float*)d_in[0];
  const float* Y   = (const float*)d_in[1];
  const float* qs  = (const float*)d_in[2];
  const float* ks_ = (const float*)d_in[3];
  const float* vs  = (const float*)d_in[4];
  const float* Wq  = (const float*)d_in[5];
  const float* bq  = (const float*)d_in[6];
  const float* g1  = (const float*)d_in[7];
  const float* b1  = (const float*)d_in[8];
  const float* Wov = (const float*)d_in[9];
  const float* bov = (const float*)d_in[10];
  const float* g2  = (const float*)d_in[11];
  const float* b2  = (const float*)d_in[12];
  const int*   msk = (const int*)d_in[13];

  char* ws = (char*)d_ws;
  float* gq   = (float*)(ws + OFF_GATES);
  float* gk   = gq + DD;
  float* gv   = gq + 2 * DD;
  bf16* Wqb   = (bf16*)(ws + OFF_WQB);
  bf16* Xb    = (bf16*)(ws + OFF_XB);
  bf16* Qb    = (bf16*)(ws + OFF_QB);
  bf16* Kb    = (bf16*)(ws + OFF_KB);
  bf16* Vb    = (bf16*)(ws + OFF_VB);
  float* qlin = (float*)(ws + OFF_QLIN);   // later reused as attention output
  float* scor = (float*)(ws + OFF_P);      // f32 scores, then bf16 p in place
  bf16*  Pb   = (bf16*)(ws + OFF_P);

  const long long nX = (long long)BB * SS * DD;   // 16M
  const long long nY = (long long)BB * TT * DD;   // 16M
  const long long nW = (long long)DD * DD;        // 1M

  // 1. gates (tiny GEMV + sigmoids)
  gates_kernel<<<DD / 256, 256, 0, stream>>>(qs, ks_, vs, Wov, bov, gq, gk, gv);

  // 2. bf16 conversions (+ fused k/v gating)
  f32_to_bf16_kernel<<<(unsigned)(nW / 256), 256, 0, stream>>>(Wq, Wqb, nW);
  f32_to_bf16_kernel<<<(unsigned)(nX / 256), 256, 0, stream>>>(X, Xb, nX);
  gate_convert_y<<<(unsigned)(nY / 256), 256, 0, stream>>>(Y, gk, gv, Kb, Vb);

  // 3. Q projection: qlin = Xb @ Wqb^T + bq   (M=16384, N=1024, K=1024, B is [N][K])
  wmma_gemm<false><<<dim3(DD / 128, (BB * SS) / 128, 1), 256, 0, stream>>>(
      Xb, Wqb, qlin, DD, DD, DD, DD, 0LL, 0LL, 0LL, bq, 1.0f);

  // 4. LayerNorm + q-gate -> bf16 q
  ln_gate_to_bf16<<<BB * SS, 256, 0, stream>>>(qlin, Qb, g1, b1, gq);

  // 5. scores = q @ k^T / sqrt(D)   per batch (M=S, N=T, K=D, B is [N][K])
  wmma_gemm<false><<<dim3(TT / 128, SS / 128, BB), 256, 0, stream>>>(
      Qb, Kb, scor, DD, DD, DD, TT,
      (long long)SS * DD, (long long)TT * DD, (long long)SS * TT,
      nullptr, 1.0f / 32.0f);

  // 6. masked softmax; writes bf16 p in place (ld = 2*T halves)
  softmax_rows<<<BB * SS, 256, 0, stream>>>(scor, msk);

  // 7. out = p @ v   per batch (M=S, N=D, K=T, B is [K][N]); reuse qlin region
  wmma_gemm<true><<<dim3(DD / 128, SS / 128, BB), 256, 0, stream>>>(
      Pb, Vb, qlin, TT, 2 * TT, DD, DD,
      (long long)SS * 2 * TT, (long long)TT * DD, (long long)SS * DD,
      nullptr, 1.0f);

  // 8. final LayerNorm(X + out) -> d_out (f32)
  add_ln_kernel<<<BB * SS, 256, 0, stream>>>(X, qlin, g2, b2, (float*)d_out);
}
